// Encoder_64544768524654
// MI455X (gfx1250) — matrix operand order repaired and checked by reading, not compiled, not run on hardware
//
#include <hip/hip_runtime.h>
#include <hip/hip_bf16.h>

typedef _Float16 f16;
typedef __attribute__((ext_vector_type(16))) _Float16 v16h;
typedef __attribute__((ext_vector_type(8)))  _Float16 v8h;
typedef __attribute__((ext_vector_type(8)))  float    v8f;

#define DIM 768
#define HEADS 12
#define DH 64
#define NSEQ 1024
#define BATCH 8
#define MLP 3072
#define MROWS (BATCH * NSEQ)            // 8192
#define QKV_STRIDE ((size_t)BATCH * HEADS * NSEQ * DH)  // 6,291,456 elems

static __device__ __forceinline__ v16h cat16(v8h lo, v8h hi) {
    return __builtin_shufflevector(lo, hi, 0,1,2,3,4,5,6,7,8,9,10,11,12,13,14,15);
}

// Async DMA: global -> LDS, 16B per lane, tracked by ASYNCcnt (CDNA5 TDM-lite path).
static __device__ __forceinline__ void async_g2l_b128(const void* gaddr, void* lds) {
    unsigned int  l = (unsigned int)(unsigned long long)(uintptr_t)lds;  // low 32b = LDS offset
    unsigned long long g = (unsigned long long)(uintptr_t)gaddr;
    asm volatile("global_load_async_to_lds_b128 %0, %1, off" :: "v"(l), "v"(g) : "memory");
}
static __device__ __forceinline__ void wait_async0() {
    asm volatile("s_wait_asynccnt 0x0" ::: "memory");
}

// ---------------------------------------------------------------- cast f32->f16
__global__ __launch_bounds__(256) void cast_f32_f16(const float* __restrict__ in,
                                                    f16* __restrict__ out, int n) {
    int i = blockIdx.x * 256 + threadIdx.x;
    if (i < n) out[i] = (f16)in[i];
}

// ---------------------------------------------------------------- LayerNorm + cast
__global__ __launch_bounds__(256) void ln_cast_kernel(const float* __restrict__ x,
                                                      const float* __restrict__ g,
                                                      const float* __restrict__ b,
                                                      f16* __restrict__ out) {
    const int row = blockIdx.x;
    const int t = threadIdx.x;
    const float* xr = x + (size_t)row * DIM;
    float v0 = xr[t], v1 = xr[t + 256], v2 = xr[t + 512];
    __shared__ float rs[256], rq[256];
    rs[t] = v0 + v1 + v2;
    rq[t] = v0 * v0 + v1 * v1 + v2 * v2;
    __syncthreads();
    for (int st = 128; st > 0; st >>= 1) {
        if (t < st) { rs[t] += rs[t + st]; rq[t] += rq[t + st]; }
        __syncthreads();
    }
    const float mu = rs[0] * (1.0f / DIM);
    const float var = rq[0] * (1.0f / DIM) - mu * mu;
    const float rst = rsqrtf(var + 1e-5f);
    f16* o = out + (size_t)row * DIM;
    o[t]       = (f16)((v0 - mu) * rst * g[t]       + b[t]);
    o[t + 256] = (f16)((v1 - mu) * rst * g[t + 256] + b[t + 256]);
    o[t + 512] = (f16)((v2 - mu) * rst * g[t + 512] + b[t + 512]);
}

// ---------------------------------------------------------------- WMMA GEMM
// C[M,N] = A[M,K] * B[K,N]; 128x128 block tile, K-step 32, 8 waves (4x2),
// wave tile 32x64 = 2x4 WMMA 16x16x32 f16->f32. Double-buffered LDS; A tile
// staged via GLOBAL_LOAD_ASYNC_TO_LDS_B128, B tile transposed via ds scatter.
enum { EPI_QKV = 0, EPI_RES = 1, EPI_GELU = 2 };

template <int EPI>
__global__ __launch_bounds__(256) void gemm_wmma(
    const f16* __restrict__ A, const f16* __restrict__ Bw,
    int M, int N, int K,
    float* __restrict__ outF, f16* __restrict__ outH,
    const float* __restrict__ bias, const float* __restrict__ res) {
    __shared__ f16 As[2][128][40];   // 80B row stride -> conflict-free 16B lanes
    __shared__ f16 Bs[2][128][40];   // B stored transposed: Bs[n][k]

    const int t = threadIdx.x;
    const int lane = t & 31, wave = t >> 5;
    const int wm = wave >> 1, wn = wave & 1;
    const int half = lane >> 4, l16 = lane & 15;
    const int row0 = blockIdx.y * 128, col0 = blockIdx.x * 128;

    v8f acc[2][4];
#pragma unroll
    for (int mi = 0; mi < 2; ++mi)
#pragma unroll
        for (int nj = 0; nj < 4; ++nj)
#pragma unroll
            for (int r = 0; r < 8; ++r) acc[mi][nj][r] = 0.0f;

    const int ktiles = K >> 5;

    // staging helper (A async-to-LDS, B sync + transpose scatter)
    auto stage = [&](int buf, int k0) {
#pragma unroll
        for (int i = 0; i < 2; ++i) {
            int chunk = t + (i << 8);
            int r = chunk >> 2, c8 = (chunk & 3) << 3;
            async_g2l_b128(&A[(size_t)(row0 + r) * K + k0 + c8], &As[buf][r][c8]);
        }
#pragma unroll
        for (int i = 0; i < 2; ++i) {
            int chunk = t + (i << 8);
            int krow = chunk >> 4, n8 = (chunk & 15) << 3;
            v8h vv = *(const v8h*)&Bw[(size_t)(k0 + krow) * N + col0 + n8];
#pragma unroll
            for (int j = 0; j < 8; ++j) Bs[buf][n8 + j][krow] = vv[j];
        }
    };

    stage(0, 0);
    wait_async0();
    __syncthreads();

    for (int kk = 0; kk < ktiles; ++kk) {
        const int cur = kk & 1;
        if (kk + 1 < ktiles) {
            stage(cur ^ 1, (kk + 1) << 5);
            if (kk + 2 < ktiles) {  // warm L2 two tiles ahead (global_prefetch_b8)
                int k2 = (kk + 2) << 5;
                __builtin_prefetch(&Bw[(size_t)(k2 + (t >> 3)) * N + col0 + ((t & 7) << 4)], 0, 3);
                __builtin_prefetch(&A[(size_t)(row0 + (t >> 1)) * K + k2 + ((t & 1) << 4)], 0, 3);
            }
        }

        v16h a[2], b[4];
#pragma unroll
        for (int mi = 0; mi < 2; ++mi) {
            const f16* p = &As[cur][wm * 32 + mi * 16 + l16][half * 8];
            a[mi] = cat16(*(const v8h*)p, *(const v8h*)(p + 16));
        }
#pragma unroll
        for (int nj = 0; nj < 4; ++nj) {
            const f16* p = &Bs[cur][wn * 64 + nj * 16 + l16][half * 8];
            b[nj] = cat16(*(const v8h*)p, *(const v8h*)(p + 16));
        }
#pragma unroll
        for (int mi = 0; mi < 2; ++mi)
#pragma unroll
            for (int nj = 0; nj < 4; ++nj)
                acc[mi][nj] = __builtin_amdgcn_wmma_f32_16x16x32_f16(
                    false, a[mi], false, b[nj], (short)0, acc[mi][nj], false, false);

        wait_async0();
        __syncthreads();
    }

    // epilogue: C layout — VGPR r, lanes 0-15 -> M=r, lanes 16-31 -> M=r+8
#pragma unroll
    for (int mi = 0; mi < 2; ++mi)
#pragma unroll
        for (int nj = 0; nj < 4; ++nj)
#pragma unroll
            for (int r = 0; r < 8; ++r) {
                const int row = row0 + wm * 32 + mi * 16 + r + half * 8;
                const int col = col0 + wn * 64 + nj * 16 + l16;
                float v = acc[mi][nj][r];
                if (EPI == EPI_QKV) {
                    const int which = col / DIM;
                    const int rem = col - which * DIM;
                    const int head = rem >> 6, d = rem & 63;
                    const int bb = row >> 10, n = row & 1023;
                    outH[(size_t)which * QKV_STRIDE +
                         (((size_t)(bb * HEADS + head) * NSEQ) + n) * DH + d] = (f16)v;
                } else if (EPI == EPI_RES) {
                    float bv = bias ? bias[col] : 0.0f;
                    outF[(size_t)row * N + col] = v + bv + res[(size_t)row * N + col];
                } else {  // EPI_GELU
                    float xg = v + bias[col];
                    float gl = 0.5f * xg *
                               (1.0f + tanhf(0.7978845608028654f *
                                             (xg + 0.044715f * xg * xg * xg)));
                    outH[(size_t)row * N + col] = (f16)gl;
                }
            }
}

// ---------------------------------------------------------------- Flash attention
// One block: 128 query rows of one (b,h); 8 waves, 16 rows/wave; 64-key tiles.
// K tile staged async-to-LDS; V transposed via ds scatter.
__global__ __launch_bounds__(256) void attn_kernel(const f16* __restrict__ q,
                                                   const f16* __restrict__ k,
                                                   const f16* __restrict__ v,
                                                   f16* __restrict__ out) {
    __shared__ f16 Ks[64][72];       // K tile row-major [key][d]
    __shared__ f16 Vt[64][72];       // V tile transposed [d][key]
    __shared__ f16 Ps[8][16][72];    // per-wave P tile [row][key]

    const int t = threadIdx.x;
    const int lane = t & 31, wave = t >> 5;
    const int half = lane >> 4, l16 = lane & 15;
    const int bh = blockIdx.x >> 3;          // b*HEADS + h
    const int qt = blockIdx.x & 7;
    const int bb = bh / HEADS, hh = bh - bb * HEADS;
    const size_t base = (size_t)bh * NSEQ * DH;
    const int rows0 = qt * 128 + wave * 16;
    const float scale = 0.125f;              // 64^-0.5

    // Q fragments (2 K-steps over d), kept in registers for whole kernel
    v16h aq[2];
#pragma unroll
    for (int f = 0; f < 2; ++f) {
        const f16* p = q + base + (size_t)(rows0 + l16) * DH + f * 32 + half * 8;
        aq[f] = cat16(*(const v8h*)p, *(const v8h*)(p + 16));
    }

    v8f o[4];
    float Mr[8], Lr[8];
#pragma unroll
    for (int j = 0; j < 4; ++j)
#pragma unroll
        for (int r = 0; r < 8; ++r) o[j][r] = 0.0f;
#pragma unroll
    for (int r = 0; r < 8; ++r) { Mr[r] = -1e30f; Lr[r] = 0.0f; }

    for (int kt = 0; kt < NSEQ / 64; ++kt) {
        const int key0 = kt * 64;
        __syncthreads();
        // stage K (async DMA to LDS) and V^T (transpose scatter), 64x64 f16 each
#pragma unroll
        for (int i = 0; i < 2; ++i) {
            int chunk = t + (i << 8);
            int kr = chunk >> 3, d8 = (chunk & 7) << 3;
            async_g2l_b128(&k[base + (size_t)(key0 + kr) * DH + d8], &Ks[kr][d8]);
            v8h vv = *(const v8h*)&v[base + (size_t)(key0 + kr) * DH + d8];
#pragma unroll
            for (int j = 0; j < 8; ++j) Vt[d8 + j][kr] = vv[j];
        }
        if (kt + 1 < NSEQ / 64) {  // prefetch next K/V tile into L2
            __builtin_prefetch(&k[base + (size_t)(key0 + 64 + (t >> 2)) * DH + ((t & 3) << 4)], 0, 3);
            __builtin_prefetch(&v[base + (size_t)(key0 + 64 + (t >> 2)) * DH + ((t & 3) << 4)], 0, 3);
        }
        wait_async0();
        __syncthreads();

        // S = Q K^T : 16x64 per wave (4 col tiles, 2 K-steps over d)
        v8f s[4];
#pragma unroll
        for (int j = 0; j < 4; ++j)
#pragma unroll
            for (int r = 0; r < 8; ++r) s[j][r] = 0.0f;
#pragma unroll
        for (int ks = 0; ks < 2; ++ks)
#pragma unroll
            for (int j = 0; j < 4; ++j) {
                const f16* p = &Ks[j * 16 + l16][ks * 32 + half * 8];
                v16h bk = cat16(*(const v8h*)p, *(const v8h*)(p + 16));
                s[j] = __builtin_amdgcn_wmma_f32_16x16x32_f16(
                    false, aq[ks], false, bk, (short)0, s[j], false, false);
            }

        // online softmax per row (row r lives on 16 lanes of one half)
#pragma unroll
        for (int r = 0; r < 8; ++r) {
            float mloc = -1e30f;
#pragma unroll
            for (int j = 0; j < 4; ++j) {
                float sv = s[j][r] * scale;
                s[j][r] = sv;
                mloc = fmaxf(mloc, sv);
            }
#pragma unroll
            for (int m = 1; m < 16; m <<= 1) mloc = fmaxf(mloc, __shfl_xor(mloc, m, 32));
            const float mnew = fmaxf(Mr[r], mloc);
            const float corr = __expf(Mr[r] - mnew);
            float sum = 0.0f;
#pragma unroll
            for (int j = 0; j < 4; ++j) {
                float pv = __expf(s[j][r] - mnew);
                s[j][r] = pv;
                sum += pv;
            }
#pragma unroll
            for (int m = 1; m < 16; m <<= 1) sum += __shfl_xor(sum, m, 32);
            Lr[r] = Lr[r] * corr + sum;
            Mr[r] = mnew;
#pragma unroll
            for (int j = 0; j < 4; ++j) o[j][r] *= corr;
        }

        // C-layout -> A-fragment layout via LDS
#pragma unroll
        for (int j = 0; j < 4; ++j)
#pragma unroll
            for (int r = 0; r < 8; ++r)
                Ps[wave][r + half * 8][j * 16 + l16] = (f16)s[j][r];
        __syncthreads();

        // O += P V : K-dim = 64 keys (2 steps), 4 d-col tiles
#pragma unroll
        for (int ks = 0; ks < 2; ++ks) {
            const f16* pp = &Ps[wave][l16][ks * 32 + half * 8];
            v16h ap = cat16(*(const v8h*)pp, *(const v8h*)(pp + 16));
#pragma unroll
            for (int j = 0; j < 4; ++j) {
                const f16* p = &Vt[j * 16 + l16][ks * 32 + half * 8];
                v16h bv = cat16(*(const v8h*)p, *(const v8h*)(p + 16));
                o[j] = __builtin_amdgcn_wmma_f32_16x16x32_f16(
                    false, ap, false, bv, (short)0, o[j], false, false);
            }
        }
    }

    // normalize and store to [B, N, H, D]
#pragma unroll
    for (int j = 0; j < 4; ++j)
#pragma unroll
        for (int r = 0; r < 8; ++r) {
            const int n = rows0 + r + half * 8;
            const int d = j * 16 + l16;
            const float val = o[j][r] * (1.0f / Lr[r]);
            out[(((size_t)(bb * NSEQ + n)) * HEADS + hh) * DH + d] = (f16)val;
        }
}

// ---------------------------------------------------------------- host
extern "C" void kernel_launch(void* const* d_in, const int* in_sizes, int n_in,
                              void* d_out, int out_size, void* d_ws, size_t ws_size,
                              hipStream_t stream) {
    const float* x     = (const float*)d_in[0];
    const float* ln1_g = (const float*)d_in[1];
    const float* ln1_b = (const float*)d_in[2];
    const float* w_qkv = (const float*)d_in[3];
    const float* w_out = (const float*)d_in[4];
    const float* ln2_g = (const float*)d_in[5];
    const float* ln2_b = (const float*)d_in[6];
    const float* w1    = (const float*)d_in[7];
    const float* b1    = (const float*)d_in[8];
    const float* w2    = (const float*)d_in[9];
    const float* b2    = (const float*)d_in[10];
    float* outp = (float*)d_out;

    char* ws = (char*)d_ws;
    size_t off = 0;
    auto take = [&](size_t bytes) { char* p = ws + off; off += (bytes + 255) & ~(size_t)255; return p; };

    f16*   h16   = (f16*)take((size_t)MROWS * DIM * 2);
    f16*   wqkvh = (f16*)take((size_t)DIM * 3 * DIM * 2);
    f16*   wouth = (f16*)take((size_t)DIM * DIM * 2);
    f16*   w1h   = (f16*)take((size_t)DIM * MLP * 2);
    f16*   w2h   = (f16*)take((size_t)MLP * DIM * 2);
    f16*   qkv16 = (f16*)take(3 * QKV_STRIDE * 2);
    f16*   attn16= (f16*)take((size_t)MROWS * DIM * 2);
    float* x1    = (float*)take((size_t)MROWS * DIM * 4);
    f16*   m16   = (f16*)take((size_t)MROWS * MLP * 2);

    // 1) cast weights to f16
    {
        int n;
        n = DIM * 3 * DIM; cast_f32_f16<<<(n + 255) / 256, 256, 0, stream>>>(w_qkv, wqkvh, n);
        n = DIM * DIM;     cast_f32_f16<<<(n + 255) / 256, 256, 0, stream>>>(w_out, wouth, n);
        n = DIM * MLP;     cast_f32_f16<<<(n + 255) / 256, 256, 0, stream>>>(w1, w1h, n);
        n = MLP * DIM;     cast_f32_f16<<<(n + 255) / 256, 256, 0, stream>>>(w2, w2h, n);
    }
    // 2) LN1
    ln_cast_kernel<<<MROWS, 256, 0, stream>>>(x, ln1_g, ln1_b, h16);
    // 3) QKV projection, scatter into [B,H,N,D] f16
    gemm_wmma<EPI_QKV><<<dim3((3 * DIM) / 128, MROWS / 128), 256, 0, stream>>>(
        h16, wqkvh, MROWS, 3 * DIM, DIM, nullptr, qkv16, nullptr, nullptr);
    // 4) attention
    attn_kernel<<<BATCH * HEADS * (NSEQ / 128), 256, 0, stream>>>(
        qkv16, qkv16 + QKV_STRIDE, qkv16 + 2 * QKV_STRIDE, attn16);
    // 5) out projection + residual -> x1 (f32)
    gemm_wmma<EPI_RES><<<dim3(DIM / 128, MROWS / 128), 256, 0, stream>>>(
        attn16, wouth, MROWS, DIM, DIM, x1, nullptr, nullptr, x);
    // 6) LN2 (reuse h16)
    ln_cast_kernel<<<MROWS, 256, 0, stream>>>(x1, ln2_g, ln2_b, h16);
    // 7) FFN up + GELU -> m16 (f16)
    gemm_wmma<EPI_GELU><<<dim3(MLP / 128, MROWS / 128), 256, 0, stream>>>(
        h16, w1h, MROWS, MLP, DIM, nullptr, m16, b1, nullptr);
    // 8) FFN down + bias + residual -> d_out (f32)
    gemm_wmma<EPI_RES><<<dim3(DIM / 128, MROWS / 128), 256, 0, stream>>>(
        m16, w2h, MROWS, DIM, MLP, outp, nullptr, b2, x1);
    (void)in_sizes; (void)n_in; (void)out_size; (void)ws_size;
}
